// SwinTransformerDecode_6004364280664
// MI455X (gfx1250) — matrix operand order repaired and checked by reading, not compiled, not run on hardware
//
#include <hip/hip_runtime.h>
#include <hip/hip_bf16.h>

// ---------------------------------------------------------------------------
// CDNA5 (gfx1250) WMMA types & helpers
// ---------------------------------------------------------------------------
typedef __bf16 v16bf __attribute__((ext_vector_type(16)));
typedef float  v8f   __attribute__((ext_vector_type(8)));

// float -> bf16, round-half-up (1 add + shift); pack pair with one v_perm_b32
static __device__ __forceinline__ unsigned short f2bf(float f) {
    unsigned u = __builtin_bit_cast(unsigned, f) + 0x8000u;
    return (unsigned short)(u >> 16);
}
static __device__ __forceinline__ unsigned pack2(float a, float b) {
    unsigned ua = __builtin_bit_cast(unsigned, a) + 0x8000u;
    unsigned ub = __builtin_bit_cast(unsigned, b) + 0x8000u;
    // result bytes: [ua.b2, ua.b3, ub.b2, ub.b3]  (S0=ub -> sel 4..7, S1=ua -> sel 0..3)
    return __builtin_amdgcn_perm(ub, ua, 0x07060302u);
}

static __device__ __forceinline__ v8f wmma_bf16(v16bf a, v16bf b, v8f c) {
    // v_wmma_f32_16x16x32_bf16
    return __builtin_amdgcn_wmma_f32_16x16x32_bf16(
        false, a, false, b, (short)0, c, false, false);
}

// A-matrix fragment (16x32 bf16, M x K), per ISA 7.12.2
static __device__ __forceinline__ v16bf load_a_frag(const unsigned short* base,
                                                    int row0, int rowStride,
                                                    int koff, int lane) {
    int m  = row0 + (lane & 15);
    int kb = ((lane < 16) ? 0 : 8) + koff;
    const unsigned* rp = (const unsigned*)(base + m * rowStride);
    union { v16bf v; unsigned u[8]; } f;
#pragma unroll
    for (int i = 0; i < 4; ++i) {
        f.u[i]     = rp[(kb >> 1) + i];
        f.u[i + 4] = rp[((kb + 16) >> 1) + i];
    }
    return f.v;
}

// B-matrix fragment (32x16 bf16, K x N), B element (k,n) = rowMajorSrc[n][k]
static __device__ __forceinline__ v16bf load_b_frag(const unsigned short* base,
                                                    int n0, int rowStride,
                                                    int koff, int lane) {
    int n  = n0 + (lane & 15);
    int kb = ((lane < 16) ? 0 : 8) + koff;
    const unsigned* rp = (const unsigned*)(base + n * rowStride);
    union { v16bf v; unsigned u[8]; } f;
#pragma unroll
    for (int i = 0; i < 4; ++i) {
        f.u[i]     = rp[(kb >> 1) + i];
        f.u[i + 4] = rp[((kb + 16) >> 1) + i];
    }
    return f.v;
}

// CDNA5 async global->LDS copy (ASYNCcnt-tracked, bypasses VGPR data path).
static __device__ __forceinline__ void async_load_b128(unsigned ldsOff,
                                                       unsigned long long gaddr) {
    asm volatile("global_load_async_to_lds_b128 %0, %1, off"
                 :: "v"(ldsOff), "v"(gaddr) : "memory");
}
static __device__ __forceinline__ void wait_asynccnt0() {
    asm volatile("s_wait_asynccnt 0" ::: "memory");
}

// ---------------------------------------------------------------------------
// Constants
// ---------------------------------------------------------------------------
#define DIMC   256
#define NTOK   64          // tokens per window (8x8)
#define HEADS  8
#define HD     32
#define NWIN   1024        // (256/8)^2
#define SCALE  0.17677669529663687f   // 32^-0.5

// ---------------------------------------------------------------------------
// Kernel 1: fused q / kv projection, bf16 staged per (window, head)
//   grid (12, 4096), block 256 (8 waves). 64 tokens x 64 features per block.
//   Software-pipelined K-loop: next global tile prefetched into registers
//   while WMMAs consume the current LDS tile.
// ---------------------------------------------------------------------------
__global__ __launch_bounds__(256)
void swin_qkv_kernel(const float* __restrict__ x, const float* __restrict__ y,
                     const float* __restrict__ q_w, const float* __restrict__ q_b,
                     const float* __restrict__ kv_w, const float* __restrict__ kv_b,
                     unsigned short* __restrict__ Qw,
                     unsigned short* __restrict__ Kw,
                     unsigned short* __restrict__ Vw)
{
    __shared__ __align__(16) unsigned short As[64 * 32];
    __shared__ __align__(16) unsigned short Bs[64 * 32];

    const int tid  = threadIdx.x;
    const int lane = tid & 31;
    const int wave = tid >> 5;
    const int f0   = blockIdx.x * 64;           // global output feature block (0..767)
    const int tok0 = blockIdx.y * 64;

    const float* A;
    const float* W;
    const float* bias;
    unsigned short* dstb;
    float sc = 1.0f;
    int fw0;
    if (f0 < 256)      { A = x; W = q_w;  bias = q_b;  fw0 = f0;       dstb = Qw; sc = SCALE; }
    else if (f0 < 512) { A = y; W = kv_w; bias = kv_b; fw0 = f0 - 256; dstb = Kw; }
    else               { A = y; W = kv_w; bias = kv_b; fw0 = f0 - 256; dstb = Vw; }
    // local feature base within the destination tensor's 256-wide space
    const int fl0 = (f0 < 256) ? f0 : ((f0 < 512) ? (f0 - 256) : (f0 - 512));

    const int mt  = (wave & 3) * 16;            // 16-row subtile
    const int nt2 = (wave >> 2) * 32;           // 32-col subtile (2 wmma tiles)

    // per-thread staging slot: 8 contiguous elements of one row
    const int e  = tid * 8;
    const int rr = e >> 5, cc = e & 31;
    const float* Abase = A + (size_t)(tok0 + rr) * DIMC + cc;
    const float* Wbase = W + (size_t)(fw0 + rr) * DIMC + cc;
    unsigned* Adst = (unsigned*)(As + rr * 32 + cc);
    unsigned* Bdst = (unsigned*)(Bs + rr * 32 + cc);

    // pipeline prologue: tile 0 into registers
    float4 ra0 = *(const float4*)(Abase + 0);
    float4 ra1 = *(const float4*)(Abase + 4);
    float4 rb0 = *(const float4*)(Wbase + 0);
    float4 rb1 = *(const float4*)(Wbase + 4);

    v8f c0 = {}; v8f c1 = {};

    for (int k0 = 0; k0 < DIMC; k0 += 32) {
        // commit current tile to LDS (fp32 -> bf16 via v_perm packing)
        Adst[0] = pack2(ra0.x, ra0.y); Adst[1] = pack2(ra0.z, ra0.w);
        Adst[2] = pack2(ra1.x, ra1.y); Adst[3] = pack2(ra1.z, ra1.w);
        Bdst[0] = pack2(rb0.x, rb0.y); Bdst[1] = pack2(rb0.z, rb0.w);
        Bdst[2] = pack2(rb1.x, rb1.y); Bdst[3] = pack2(rb1.z, rb1.w);
        __syncthreads();

        v16bf a  = load_a_frag(As, mt, 32, 0, lane);
        v16bf b0 = load_b_frag(Bs, nt2,      32, 0, lane);
        v16bf b1 = load_b_frag(Bs, nt2 + 16, 32, 0, lane);

        // issue next tile's global loads before the WMMAs (latency hidden)
        if (k0 + 32 < DIMC) {
            ra0 = *(const float4*)(Abase + k0 + 32);
            ra1 = *(const float4*)(Abase + k0 + 36);
            rb0 = *(const float4*)(Wbase + k0 + 32);
            rb1 = *(const float4*)(Wbase + k0 + 36);
            if (k0 + 64 < DIMC) __builtin_prefetch(Abase + k0 + 64, 0, 0);
        }

        c0 = wmma_bf16(a, b0, c0);
        c1 = wmma_bf16(a, b1, c1);
        __syncthreads();
    }

    // Epilogue: scatter into per-(window, head) bf16 tiles (region uniform per block)
    const int half = lane >> 4, n = lane & 15;
#pragma unroll
    for (int r = 0; r < 8; ++r) {
        int row  = tok0 + mt + r + half * 8;
        int bwin = row >> 6, ntok = row & 63;
        size_t rowbase = (((size_t)bwin * HEADS) * NTOK + ntok) * HD;
#pragma unroll
        for (int t = 0; t < 2; ++t) {
            float v = (t == 0) ? c0[r] : c1[r];
            int fl   = fl0 + nt2 + t * 16 + n;        // 0..255 within q/k/v
            int bidx = (f0 >= 512) ? (fl + 256) : fl; // kv_b packs [k | v]
            v = (v + bias[bidx]) * sc;
            int h = fl >> 5, d = fl & 31;
            dstb[rowbase + (size_t)h * (NTOK * HD) + d] = f2bf(v);
        }
    }
}

// ---------------------------------------------------------------------------
// Kernel 2: windowed attention, one wave per (window, head)
//   grid (8, 4096), block 32
// ---------------------------------------------------------------------------
__global__ __launch_bounds__(32)
void swin_attn_kernel(const unsigned short* __restrict__ Qw,
                      const unsigned short* __restrict__ Kw,
                      const unsigned short* __restrict__ Vw,
                      const float* __restrict__ mask,
                      const float* __restrict__ rpb,
                      unsigned short* __restrict__ Ow)
{
    __shared__ __align__(16) unsigned short Qs[NTOK * HD];
    __shared__ __align__(16) unsigned short Ks[NTOK * HD];
    __shared__ __align__(16) unsigned short Vt[HD * NTOK];   // transposed: [d][j]
    __shared__ __align__(16) unsigned short Ps[NTOK * NTOK];

    const int lane = threadIdx.x;
    const int h    = blockIdx.x;
    const int b    = blockIdx.y;
    const size_t base = ((size_t)b * HEADS + h) * (NTOK * HD);

    {   // stage Q, K via CDNA5 async global->LDS (ASYNCcnt), V transposed manually
        unsigned qOff = (unsigned)(uintptr_t)(void*)Qs;
        unsigned kOff = (unsigned)(uintptr_t)(void*)Ks;
        unsigned long long qAddr = (unsigned long long)(uintptr_t)(Qw + base);
        unsigned long long kAddr = (unsigned long long)(uintptr_t)(Kw + base);
#pragma unroll
        for (int i = 0; i < 8; ++i) {
            unsigned boff = (unsigned)(lane + i * 32) * 16u;   // 16B per lane-slot
            async_load_b128(qOff + boff, qAddr + boff);
            async_load_b128(kOff + boff, kAddr + boff);
        }
        // overlap: V transpose-scatter while async copies are in flight
        const unsigned short* vs = Vw + base;
        for (int e = lane; e < NTOK * HD; e += 32)
            Vt[(e & 31) * NTOK + (e >> 5)] = vs[e];
        wait_asynccnt0();
    }
    __syncthreads();

    const float* mrow = mask + (size_t)(b & (NWIN - 1)) * (NTOK * NTOK);
    const int half = lane >> 4, n = lane & 15;

    for (int it = 0; it < 4; ++it) {
        v16bf aQ = load_a_frag(Qs, it * 16, HD, 0, lane);
        v8f s[4];
#pragma unroll
        for (int jt = 0; jt < 4; ++jt) {
            v16bf bK = load_b_frag(Ks, jt * 16, HD, 0, lane);
            v8f z = {};
            s[jt] = wmma_bf16(aQ, bK, z);
        }
        // + relative position bias + shifted-window mask
#pragma unroll
        for (int r = 0; r < 8; ++r) {
            int row = it * 16 + r + half * 8;
            int yi = row >> 3, xi = row & 7;
#pragma unroll
            for (int jt = 0; jt < 4; ++jt) {
                int col = jt * 16 + n;
                int yj = col >> 3, xj = col & 7;
                int idx = (yi - yj + 7) * 15 + (xi - xj + 7);
                s[jt][r] += rpb[idx * HEADS + h] + mrow[row * NTOK + col];
            }
        }
        // softmax per row (row lives across 16 lanes of one half-wave)
#pragma unroll
        for (int r = 0; r < 8; ++r) {
            float mx = fmaxf(fmaxf(s[0][r], s[1][r]), fmaxf(s[2][r], s[3][r]));
            for (int off = 1; off < 16; off <<= 1)
                mx = fmaxf(mx, __shfl_xor(mx, off, 32));
            float sum = 0.f;
#pragma unroll
            for (int jt = 0; jt < 4; ++jt) {
                float e = __expf(s[jt][r] - mx);
                s[jt][r] = e; sum += e;
            }
            for (int off = 1; off < 16; off <<= 1)
                sum += __shfl_xor(sum, off, 32);
            float inv = 1.0f / sum;
#pragma unroll
            for (int jt = 0; jt < 4; ++jt) s[jt][r] *= inv;
        }
        // P -> LDS (bf16, row-major), then O_i = P_i @ V
#pragma unroll
        for (int r = 0; r < 8; ++r) {
            int row = it * 16 + r + half * 8;
#pragma unroll
            for (int jt = 0; jt < 4; ++jt)
                Ps[row * NTOK + jt * 16 + n] = f2bf(s[jt][r]);
        }
        v8f o0 = {}, o1 = {};
#pragma unroll
        for (int kk = 0; kk < NTOK; kk += 32) {
            v16bf aP  = load_a_frag(Ps, it * 16, NTOK, kk, lane);
            v16bf bV0 = load_b_frag(Vt, 0,  NTOK, kk, lane);
            v16bf bV1 = load_b_frag(Vt, 16, NTOK, kk, lane);
            o0 = wmma_bf16(aP, bV0, o0);
            o1 = wmma_bf16(aP, bV1, o1);
        }
#pragma unroll
        for (int r = 0; r < 8; ++r) {
            int row = it * 16 + r + half * 8;
            size_t off = ((size_t)b * NTOK + row) * DIMC + h * HD;
            Ow[off + n]      = f2bf(o0[r]);
            Ow[off + 16 + n] = f2bf(o1[r]);
        }
    }
}

// ---------------------------------------------------------------------------
// Kernel 3: output projection (bf16 O @ proj_w^T + proj_b -> fp32)
//   grid (4, 4096), block 256; software-pipelined like kernel 1.
// ---------------------------------------------------------------------------
__global__ __launch_bounds__(256)
void swin_proj_kernel(const unsigned short* __restrict__ Ow,
                      const float* __restrict__ proj_w,
                      const float* __restrict__ proj_b,
                      float* __restrict__ out)
{
    __shared__ __align__(16) unsigned short As[64 * 32];
    __shared__ __align__(16) unsigned short Bs[64 * 32];

    const int tid  = threadIdx.x;
    const int lane = tid & 31;
    const int wave = tid >> 5;
    const int f0   = blockIdx.x * 64;
    const int tok0 = blockIdx.y * 64;
    const int mt   = (wave & 3) * 16;
    const int nt2  = (wave >> 2) * 32;

    const int e  = tid * 8;
    const int rr = e >> 5, cc = e & 31;
    const unsigned short* Abase = Ow + (size_t)(tok0 + rr) * DIMC + cc;
    const float* Wbase = proj_w + (size_t)(f0 + rr) * DIMC + cc;
    uint4*    Adst = (uint4*)(As + rr * 32 + cc);
    unsigned* Bdst = (unsigned*)(Bs + rr * 32 + cc);

    // pipeline prologue
    uint4  raq = *(const uint4*)(Abase + 0);
    float4 rb0 = *(const float4*)(Wbase + 0);
    float4 rb1 = *(const float4*)(Wbase + 4);

    v8f c0 = {}, c1 = {};

    for (int k0 = 0; k0 < DIMC; k0 += 32) {
        Adst[0] = raq;                               // already bf16
        Bdst[0] = pack2(rb0.x, rb0.y); Bdst[1] = pack2(rb0.z, rb0.w);
        Bdst[2] = pack2(rb1.x, rb1.y); Bdst[3] = pack2(rb1.z, rb1.w);
        __syncthreads();

        v16bf a  = load_a_frag(As, mt, 32, 0, lane);
        v16bf b0 = load_b_frag(Bs, nt2,      32, 0, lane);
        v16bf b1 = load_b_frag(Bs, nt2 + 16, 32, 0, lane);

        if (k0 + 32 < DIMC) {
            raq = *(const uint4*)(Abase + k0 + 32);
            rb0 = *(const float4*)(Wbase + k0 + 32);
            rb1 = *(const float4*)(Wbase + k0 + 36);
            if (k0 + 64 < DIMC) __builtin_prefetch(Wbase + k0 + 64, 0, 0);
        }

        c0 = wmma_bf16(a, b0, c0);
        c1 = wmma_bf16(a, b1, c1);
        __syncthreads();
    }

    const int half = lane >> 4, n = lane & 15;
#pragma unroll
    for (int r = 0; r < 8; ++r) {
        int row = tok0 + mt + r + half * 8;
        int fA  = f0 + nt2 + n;
        int fB  = fA + 16;
        out[(size_t)row * DIMC + fA] = c0[r] + proj_b[fA];
        out[(size_t)row * DIMC + fB] = c1[r] + proj_b[fB];
    }
}

// ---------------------------------------------------------------------------
// Launch
// ---------------------------------------------------------------------------
extern "C" void kernel_launch(void* const* d_in, const int* in_sizes, int n_in,
                              void* d_out, int out_size, void* d_ws, size_t ws_size,
                              hipStream_t stream) {
    const float* x      = (const float*)d_in[0];
    const float* y      = (const float*)d_in[1];
    const float* mask   = (const float*)d_in[2];
    const float* q_w    = (const float*)d_in[3];
    const float* q_b    = (const float*)d_in[4];
    const float* kv_w   = (const float*)d_in[5];
    const float* kv_b   = (const float*)d_in[6];
    const float* proj_w = (const float*)d_in[7];
    const float* proj_b = (const float*)d_in[8];
    const float* rpb    = (const float*)d_in[9];
    float* out = (float*)d_out;

    // bf16 workspace: Q, K, V per (window, head) + O token-major
    const size_t seg = (size_t)4096 * HEADS * NTOK * HD;   // 64Mi elems = 128 MB each
    unsigned short* Qw = (unsigned short*)d_ws;
    unsigned short* Kw = Qw + seg;
    unsigned short* Vw = Kw + seg;
    unsigned short* Ow = Vw + seg;

    swin_qkv_kernel<<<dim3(12, 4096), 256, 0, stream>>>(x, y, q_w, q_b, kv_w, kv_b,
                                                        Qw, Kw, Vw);
    swin_attn_kernel<<<dim3(8, 4096), 32, 0, stream>>>(Qw, Kw, Vw, mask, rpb, Ow);
    swin_proj_kernel<<<dim3(4, 4096), 256, 0, stream>>>(Ow, proj_w, proj_b, out);
}
